// NAS_SLSTM_979252543987
// MI455X (gfx1250) — matrix operand order repaired and checked by reading, not compiled, not run on hardware
//
#include <hip/hip_runtime.h>
#include <hip/hip_bf16.h>

typedef __bf16 bf16;
typedef __attribute__((ext_vector_type(8)))  __bf16 v8bf;
typedef __attribute__((ext_vector_type(16))) __bf16 v16bf;
typedef __attribute__((ext_vector_type(8)))  float  v8f;

#define B_ 128
#define F_ 1024
#define T_ 256
#define H_ 1024
#define NG 4096  /* 4*H */

// ---------------------------------------------------------------------------
// WMMA fragment loaders (CDNA5 wave32, 16-bit operand layouts per ISA 7.12.2)
// ---------------------------------------------------------------------------
// A: 16x32 (MxK), row-major source [M, ld]. lane<16: elems 0..7 -> K=kc..kc+7,
// elems 8..15 -> K=kc+16..kc+23; lanes>=16 shifted by +8 in K.
__device__ __forceinline__ v16bf load_a_frag(const bf16* __restrict__ base,
                                             int ld, int m0, int kc, int lane) {
  const int lane16 = lane & 15;
  const int half8  = (lane >> 4) << 3;  // 0 or 8
  const bf16* row = base + (size_t)(m0 + lane16) * ld + kc + half8;
  v8bf lo = *reinterpret_cast<const v8bf*>(row);        // K = kc+half8 .. +7
  v8bf hi = *reinterpret_cast<const v8bf*>(row + 16);   // K = kc+16+half8 ..
  return __builtin_shufflevector(lo, hi, 0,1,2,3,4,5,6,7,8,9,10,11,12,13,14,15);
}

// B: 32x16 (KxN), source pre-transposed row-major [N, ld] (i.e. BT[n][k]).
// lane<16: elems 0..7 -> K=kc..kc+7, elems 8..15 -> K=kc+16..kc+23; lanes>=16 +8 in K ; N = n0 + lane%16.
__device__ __forceinline__ v16bf load_b_frag(const bf16* __restrict__ baseT,
                                             int ld, int n0, int kc, int lane) {
  const int lane16 = lane & 15;
  const int half16 = (lane >> 4) << 3;  // 0 or 8
  const bf16* p = baseT + (size_t)(n0 + lane16) * ld + kc + half16;
  v8bf lo = *reinterpret_cast<const v8bf*>(p);
  v8bf hi = *reinterpret_cast<const v8bf*>(p + 16);
  return __builtin_shufflevector(lo, hi, 0,1,2,3,4,5,6,7,8,9,10,11,12,13,14,15);
}

__device__ __forceinline__ float sigmoid_fast(float x) {
  return 1.0f / (1.0f + __expf(-x));
}
__device__ __forceinline__ float tanh_fast(float x) {
  x = fminf(15.0f, fmaxf(-15.0f, x));
  float e = __expf(2.0f * x);
  return (e - 1.0f) / (e + 1.0f);
}

#define WMMA_BF16(a, b, c) \
  __builtin_amdgcn_wmma_f32_16x16x32_bf16(false, (a), false, (b), (short)0, (c), false, false)

// ---------------------------------------------------------------------------
// Prep: x [B,F,T] f32  ->  X [T,B,F] bf16   (LDS-tiled transpose+convert)
// ---------------------------------------------------------------------------
__global__ __launch_bounds__(256) void transpose_x_kernel(
    const float* __restrict__ x, bf16* __restrict__ X) {
  __shared__ float tile[32][33];
  const int t0 = blockIdx.x * 32, f0 = blockIdx.y * 32, b = blockIdx.z;
  const int tx = threadIdx.x, ty = threadIdx.y;
#pragma unroll
  for (int k = 0; k < 4; ++k)
    tile[ty + k * 8][tx] =
        x[(size_t)b * F_ * T_ + (size_t)(f0 + ty + k * 8) * T_ + t0 + tx];
  __syncthreads();
#pragma unroll
  for (int k = 0; k < 4; ++k)
    X[(size_t)(t0 + ty + k * 8) * B_ * F_ + (size_t)b * F_ + f0 + tx] =
        (bf16)tile[tx][ty + k * 8];
}

// Generic f32 [R,C] -> bf16 [C,R] transpose+convert (for W*, U*)
__global__ __launch_bounds__(256) void transpose_w_kernel(
    const float* __restrict__ src, bf16* __restrict__ dst, int R, int C) {
  __shared__ float tile[32][33];
  const int bx = blockIdx.x * 32, by = blockIdx.y * 32;
  const int tx = threadIdx.x, ty = threadIdx.y;
#pragma unroll
  for (int k = 0; k < 4; ++k)
    tile[ty + k * 8][tx] = src[(size_t)(by + ty + k * 8) * C + bx + tx];
  __syncthreads();
#pragma unroll
  for (int k = 0; k < 4; ++k)
    dst[(size_t)(bx + ty + k * 8) * R + by + tx] = (bf16)tile[tx][ty + k * 8];
}

__global__ void init_state_kernel(bf16* __restrict__ h, float* __restrict__ c, int n) {
  int i = blockIdx.x * blockDim.x + threadIdx.x;
  if (i < n) { h[i] = (bf16)0.0f; c[i] = 0.0f; }
}

// ---------------------------------------------------------------------------
// Big input projection: Z[M=T*B, 4H] = X[M,F] @ WtT + bias   (bf16 WMMA)
// Each wave: one 16-row M tile x four consecutive 16-col N tiles.
// Register double-buffered pipeline: next k-chunk's fragments are loaded
// before the current chunk's WMMAs so waits become partial, not full drains.
// ---------------------------------------------------------------------------
__global__ __launch_bounds__(256) void pregemm_kernel(
    const bf16* __restrict__ X,    // [T*B, F]
    const bf16* __restrict__ Wt,   // [4H, F]  (Wt[g*H+n][k] = Wg[k][n])
    const float* __restrict__ bfv, const float* __restrict__ biv,
    const float* __restrict__ bcv, const float* __restrict__ bov,
    float* __restrict__ Z)         // [T*B, 4H]
{
  const int lane  = threadIdx.x & 31;
  const int wave  = threadIdx.x >> 5;
  const int gw    = blockIdx.x * 8 + wave;     // 131072 waves
  const int m0    = (gw >> 6) * 16;            // 2048 M tiles
  const int cb    = (gw & 63) * 64;            // 64-col group (never crosses gate)
  const int lane16 = lane & 15;
  const int half8  = (lane >> 4) << 3;

  v8f acc0, acc1, acc2, acc3;
  {
    v8f* accs[4] = {&acc0, &acc1, &acc2, &acc3};
#pragma unroll
    for (int j = 0; j < 4; ++j) {
      const int col = cb + j * 16 + lane16;
      const int g = col >> 10, n = col & (H_ - 1);
      const float bv =
          (g == 0) ? bfv[n] : (g == 1) ? biv[n] : (g == 2) ? bcv[n] : bov[n];
#pragma unroll
      for (int r = 0; r < 8; ++r) (*accs[j])[r] = bv;
    }
  }

  // pipeline prologue
  v16bf a  = load_a_frag(X, F_, m0, 0, lane);
  v16bf b0 = load_b_frag(Wt, F_, cb + 0,  0, lane);
  v16bf b1 = load_b_frag(Wt, F_, cb + 16, 0, lane);
  v16bf b2 = load_b_frag(Wt, F_, cb + 32, 0, lane);
  v16bf b3 = load_b_frag(Wt, F_, cb + 48, 0, lane);

  for (int kc = 0; kc < F_ - 32; kc += 32) {
    const int kn = kc + 32;
    // issue next chunk's loads first (overlap with WMMAs below)
    v16bf an  = load_a_frag(X, F_, m0, kn, lane);
    v16bf bn0 = load_b_frag(Wt, F_, cb + 0,  kn, lane);
    v16bf bn1 = load_b_frag(Wt, F_, cb + 16, kn, lane);
    v16bf bn2 = load_b_frag(Wt, F_, cb + 32, kn, lane);
    v16bf bn3 = load_b_frag(Wt, F_, cb + 48, kn, lane);

    acc0 = WMMA_BF16(a, b0, acc0);
    acc1 = WMMA_BF16(a, b1, acc1);
    acc2 = WMMA_BF16(a, b2, acc2);
    acc3 = WMMA_BF16(a, b3, acc3);

    a = an; b0 = bn0; b1 = bn1; b2 = bn2; b3 = bn3;
  }
  // epilogue
  acc0 = WMMA_BF16(a, b0, acc0);
  acc1 = WMMA_BF16(a, b1, acc1);
  acc2 = WMMA_BF16(a, b2, acc2);
  acc3 = WMMA_BF16(a, b3, acc3);

#pragma unroll
  for (int r = 0; r < 8; ++r) {
    const size_t row = (size_t)(m0 + r + half8) * NG + cb + lane16;
    Z[row + 0]  = acc0[r];
    Z[row + 16] = acc1[r];
    Z[row + 32] = acc2[r];
    Z[row + 48] = acc3[r];
  }
}

// ---------------------------------------------------------------------------
// One recurrence step: acc_g = Z_t + h@U_g ; gate math ; write c, h, hs.
// Wave computes the SAME 16-col block for all 4 gates -> gate fusion is pure
// per-lane VALU (layouts align). h double-buffered across launches.
// Same register-pipelined K loop as the pre-GEMM.
// ---------------------------------------------------------------------------
__global__ __launch_bounds__(256) void step_kernel(
    const bf16* __restrict__ hprev,  // [B, H] bf16
    const bf16* __restrict__ Ut,     // [4H, H] (Ut[g*H+n][k] = Ug[k][n])
    const float* __restrict__ Zt,    // [B, 4H] (this step's slice)
    float* __restrict__ cbuf,        // [B, H]
    float* __restrict__ hs_t,        // [B, H] (this step's slice of hs)
    bf16* __restrict__ hnext)        // [B, H] bf16
{
  const int lane  = threadIdx.x & 31;
  const int wave  = threadIdx.x >> 5;
  const int gw    = blockIdx.x * 8 + wave;   // 512 waves
  const int m0    = (gw >> 6) * 16;          // 8 M tiles
  const int n0    = (gw & 63) * 16;          // 64 col blocks
  const int lane16 = lane & 15;
  const int half8  = (lane >> 4) << 3;

  v8f accF, accI, accC, accO;
#pragma unroll
  for (int r = 0; r < 8; ++r) {
    const size_t row = (size_t)(m0 + r + half8) * NG + n0 + lane16;
    accF[r] = Zt[row + 0 * H_];
    accI[r] = Zt[row + 1 * H_];
    accC[r] = Zt[row + 2 * H_];
    accO[r] = Zt[row + 3 * H_];
  }

  const bf16* U0 = Ut;
  const bf16* U1 = Ut + 1 * (size_t)H_ * H_;
  const bf16* U2 = Ut + 2 * (size_t)H_ * H_;
  const bf16* U3 = Ut + 3 * (size_t)H_ * H_;

  // pipeline prologue
  v16bf a  = load_a_frag(hprev, H_, m0, 0, lane);
  v16bf b0 = load_b_frag(U0, H_, n0, 0, lane);
  v16bf b1 = load_b_frag(U1, H_, n0, 0, lane);
  v16bf b2 = load_b_frag(U2, H_, n0, 0, lane);
  v16bf b3 = load_b_frag(U3, H_, n0, 0, lane);

  for (int kc = 0; kc < H_ - 32; kc += 32) {
    const int kn = kc + 32;
    v16bf an  = load_a_frag(hprev, H_, m0, kn, lane);
    v16bf bn0 = load_b_frag(U0, H_, n0, kn, lane);
    v16bf bn1 = load_b_frag(U1, H_, n0, kn, lane);
    v16bf bn2 = load_b_frag(U2, H_, n0, kn, lane);
    v16bf bn3 = load_b_frag(U3, H_, n0, kn, lane);

    accF = WMMA_BF16(a, b0, accF);
    accI = WMMA_BF16(a, b1, accI);
    accC = WMMA_BF16(a, b2, accC);
    accO = WMMA_BF16(a, b3, accO);

    a = an; b0 = bn0; b1 = bn1; b2 = bn2; b3 = bn3;
  }
  accF = WMMA_BF16(a, b0, accF);
  accI = WMMA_BF16(a, b1, accI);
  accC = WMMA_BF16(a, b2, accC);
  accO = WMMA_BF16(a, b3, accO);

#pragma unroll
  for (int r = 0; r < 8; ++r) {
    const size_t idx = (size_t)(m0 + r + half8) * H_ + n0 + lane16;
    const float ft = sigmoid_fast(accF[r]);
    const float it = sigmoid_fast(accI[r]);
    const float cc = tanh_fast(accC[r]);
    const float zo = accO[r];                 // no sigmoid on o in this variant
    const float cn = ft * cbuf[idx] + it * cc;
    cbuf[idx] = cn;
    const float hn = zo * tanh_fast(cn) + zo; // non-standard output formula
    hs_t[idx]  = hn;
    hnext[idx] = (bf16)hn;
  }
}

// ---------------------------------------------------------------------------
// Final: hs [T,B,H] f32 -> out [B,H,T] f32 (LDS-tiled, coalesced both ways)
// ---------------------------------------------------------------------------
__global__ __launch_bounds__(256) void transpose_out_kernel(
    const float* __restrict__ hs, float* __restrict__ out) {
  __shared__ float tile[32][33];
  const int t0 = blockIdx.x * 32, n0 = blockIdx.y * 32, b = blockIdx.z;
  const int tx = threadIdx.x, ty = threadIdx.y;
#pragma unroll
  for (int k = 0; k < 4; ++k)
    tile[ty + k * 8][tx] =
        hs[(size_t)(t0 + ty + k * 8) * B_ * H_ + (size_t)b * H_ + n0 + tx];
  __syncthreads();
#pragma unroll
  for (int k = 0; k < 4; ++k)
    out[(size_t)b * H_ * T_ + (size_t)(n0 + ty + k * 8) * T_ + t0 + tx] =
        tile[tx][ty + k * 8];
}

// ---------------------------------------------------------------------------
extern "C" void kernel_launch(void* const* d_in, const int* in_sizes, int n_in,
                              void* d_out, int out_size, void* d_ws, size_t ws_size,
                              hipStream_t stream) {
  const float* x = (const float*)d_in[0];
  const float* W[4] = {(const float*)d_in[1], (const float*)d_in[2],
                       (const float*)d_in[3], (const float*)d_in[4]};  // f,i,c,o
  const float* U[4] = {(const float*)d_in[5], (const float*)d_in[6],
                       (const float*)d_in[7], (const float*)d_in[8]};  // f,i,c,o
  const float* bfv = (const float*)d_in[9];
  const float* biv = (const float*)d_in[10];
  const float* bcv = (const float*)d_in[11];
  const float* bov = (const float*)d_in[12];
  float* out = (float*)d_out;

  char* ws = (char*)d_ws;
  size_t off = 0;
  bf16*  Xbf = (bf16*)(ws + off);  off += (size_t)T_ * B_ * F_ * 2;   // 64 MB
  bf16*  Wt  = (bf16*)(ws + off);  off += (size_t)4 * H_ * F_ * 2;    //  8 MB
  bf16*  Ut  = (bf16*)(ws + off);  off += (size_t)4 * H_ * H_ * 2;    //  8 MB
  float* Z   = (float*)(ws + off); off += (size_t)T_ * B_ * NG * 4;   // 512 MB
  float* hs  = (float*)(ws + off); off += (size_t)T_ * B_ * H_ * 4;   // 128 MB
  bf16*  hA  = (bf16*)(ws + off);  off += (size_t)B_ * H_ * 2;
  bf16*  hB  = (bf16*)(ws + off);  off += (size_t)B_ * H_ * 2;
  float* cB  = (float*)(ws + off); off += (size_t)B_ * H_ * 4;

  // 1) layout prep (transpose + fp32->bf16)
  transpose_x_kernel<<<dim3(T_ / 32, F_ / 32, B_), dim3(32, 8), 0, stream>>>(x, Xbf);
  for (int g = 0; g < 4; ++g) {
    transpose_w_kernel<<<dim3(H_ / 32, F_ / 32), dim3(32, 8), 0, stream>>>(
        W[g], Wt + (size_t)g * H_ * F_, F_, H_);
    transpose_w_kernel<<<dim3(H_ / 32, H_ / 32), dim3(32, 8), 0, stream>>>(
        U[g], Ut + (size_t)g * H_ * H_, H_, H_);
  }
  init_state_kernel<<<(B_ * H_ + 255) / 256, 256, 0, stream>>>(hA, cB, B_ * H_);

  // 2) hoisted input projection: Z = X @ Wcat + bias (one big WMMA GEMM)
  pregemm_kernel<<<16384, 256, 0, stream>>>(Xbf, Wt, bfv, biv, bcv, bov, Z);

  // 3) serial recurrence: one fused WMMA+gates kernel per step (double-buffered h)
  const bf16* hr = hA;
  bf16* hw = hB;
  for (int t = 0; t < T_; ++t) {
    step_kernel<<<64, 256, 0, stream>>>(hr, Ut, Z + (size_t)t * B_ * NG, cB,
                                        hs + (size_t)t * B_ * H_, hw);
    const bf16* tmp = hw; hw = (bf16*)hr; hr = tmp;
  }

  // 4) hs [T,B,H] -> out [B,H,T]
  transpose_out_kernel<<<dim3(T_ / 32, H_ / 32, B_), dim3(32, 8), 0, stream>>>(hs, out);
}
